// GATNode_30399778521238
// MI455X (gfx1250) — matrix operand order repaired and checked by reading, not compiled, not run on hardware
//
#include <hip/hip_runtime.h>
#include <hip/hip_bf16.h>

typedef __bf16 bf16_t;
typedef __bf16 v16bf __attribute__((ext_vector_type(16)));
typedef __bf16 v8bf  __attribute__((ext_vector_type(8)));
typedef float  v8f   __attribute__((ext_vector_type(8)));

#define ALPHA 0.2f
constexpr int B = 4, N = 2048, F = 256, T = 8, H = 4, D = 64;
constexpr int FT = F + T; // 264

// ---- K0a: vq[h,f] = Wq[h,f,:] . a1[h,:],  vk[h,f] = Wk[h,f,:] . a2[h,:]
__global__ void k_vqvk(const float* __restrict__ Wq, const float* __restrict__ Wk,
                       const float* __restrict__ a, float* __restrict__ vq,
                       float* __restrict__ vk) {
  int t = blockIdx.x * blockDim.x + threadIdx.x;
  if (t >= H * FT) return;
  int h = t / FT, f = t % FT;
  const float* ar = a + h * 2 * D;
  const float* wq = Wq + (size_t)(h * FT + f) * D;
  const float* wk = Wk + (size_t)(h * FT + f) * D;
  float sq = 0.f, sk = 0.f;
  for (int d = 0; d < D; ++d) { sq += wq[d] * ar[d]; sk += wk[d] * ar[D + d]; }
  vq[t] = sq; vk[t] = sk;
}

// ---- K0b: Wt[h][d][f] = (bf16) W[h][f][d]   (transposed, WMMA-A-friendly)
__global__ void k_wt(const float* __restrict__ W, bf16_t* __restrict__ Wt) {
  int idx = blockIdx.x * 256 + threadIdx.x; // H*D*F = 65536
  int f = idx & (F - 1); int d = (idx >> 8) & (D - 1); int h = idx >> 14;
  Wt[idx] = (bf16_t)W[((size_t)(h * F + f)) * D + d];
}

// ---- K0c: xbf = (bf16) x
__global__ void k_xbf(const float* __restrict__ x, bf16_t* __restrict__ xb) {
  size_t idx = (size_t)blockIdx.x * 256 + threadIdx.x;
  xb[idx] = (bf16_t)x[idx];
}

// ---- K1: e1[b,h,n] = h_t . vq[h],  e2 likewise with vk
__global__ void k_e12(const float* __restrict__ x, const float* __restrict__ oneh,
                      const float* __restrict__ vq, const float* __restrict__ vk,
                      float* __restrict__ e1, float* __restrict__ e2) {
  int idx = blockIdx.x * 256 + threadIdx.x; // B*H*N = 32768
  int n = idx & (N - 1); int h = (idx >> 11) & (H - 1); int b = idx >> 13;
  const float* xr = x + ((size_t)b * N + n) * F;
  const float* vqh = vq + h * FT; const float* vkh = vk + h * FT;
  float s1 = 0.f, s2 = 0.f;
  for (int f = 0; f < F; ++f) { float xv = xr[f]; s1 += xv * vqh[f]; s2 += xv * vkh[f]; }
  const float* oh = oneh + ((size_t)b * N + n) * T;
  for (int t = 0; t < T; ++t) { float ov = oh[t]; s1 += ov * vqh[F + t]; s2 += ov * vkh[F + t]; }
  e1[idx] = s1; e2[idx] = s2;
}

// ---- K2: denom[b,h,j] = sum_i adj[b,i,j] * exp(lrelu(e1[b,h,i]+e2[b,h,j]))
__global__ void k_denom(const int* __restrict__ adj, const float* __restrict__ e1,
                        const float* __restrict__ e2, float* __restrict__ denom) {
  int bid = blockIdx.x;                 // B * 8 jblocks * 16 ichunks = 512
  int b = bid >> 7; int rem = bid & 127; int jb = rem & 7; int ic = rem >> 3;
  int j = jb * 256 + threadIdx.x;
  float e2v[H];
#pragma unroll
  for (int h = 0; h < H; ++h) e2v[h] = e2[((size_t)(b * H + h)) * N + j];
  float acc[H] = {0.f, 0.f, 0.f, 0.f};
  int i0 = ic * 128;
  const int* ap = adj + ((size_t)b * N + i0) * N + j;
  for (int ii = 0; ii < 128; ++ii) {
    int av = ap[(size_t)ii * N];
#pragma unroll
    for (int h = 0; h < H; ++h) {
      float v = e1[((size_t)(b * H + h)) * N + i0 + ii] + e2v[h];
      v = v > 0.f ? v : ALPHA * v;
      acc[h] += (av != 0) ? __expf(v) : 0.f;
    }
  }
#pragma unroll
  for (int h = 0; h < H; ++h)
    atomicAdd(&denom[((size_t)(b * H + h)) * N + j], acc[h]);
}

// ---- K3: whn[b][h][d][j] = (bf16)( (x[b,j,:] @ W[h,:,d]) / denom[b,h,j] )
// GEMM via WMMA: M=d (A = W^T, bf16), N=j (B = x^T, bf16), K=f.
__global__ void k_whn(const bf16_t* __restrict__ xb, const bf16_t* __restrict__ Wt,
                      const float* __restrict__ denom, bf16_t* __restrict__ whn) {
  int bid = blockIdx.x;                 // B*H*(N/64) = 512, 4 waves/block
  int b = bid >> 7; int rem = bid & 127; int h = rem >> 5; int jg = rem & 31;
  int wave = threadIdx.x >> 5; int lane = threadIdx.x & 31;
  int n = lane & 15; int hf = (lane >> 4) & 1;
  int j0 = jg * 64 + wave * 16;
  v8f acc[4] = {};
  const bf16_t* xr = xb + ((size_t)(b * N) + j0 + n) * F;
#pragma unroll
  for (int kc = 0; kc < 8; ++kc) {
    int f0 = kc * 32;
    v16bf bvec = *(const v16bf*)(xr + f0 + 16 * hf);      // B: K contiguous per lane
#pragma unroll
    for (int dt = 0; dt < 4; ++dt) {
      const bf16_t* wr = Wt + ((size_t)(h * D + dt * 16 + n)) * F;
      v8bf lo = *(const v8bf*)(wr + f0 + 16 * hf);        // A: K = 16*hf + 0..7
      v8bf hi = *(const v8bf*)(wr + f0 + 16 * hf + 8);    // A: K = 16*hf + 8..15
      v16bf av;
#pragma unroll
      for (int q = 0; q < 8; ++q) { av[q] = lo[q]; av[q + 8] = hi[q]; }
      acc[dt] = __builtin_amdgcn_wmma_f32_16x16x32_bf16(
          false, av, false, bvec, (short)0, acc[dt], false, false);
    }
  }
  float inv = 1.f / (denom[((size_t)(b * H + h)) * N + j0 + n] + 1e-20f);
#pragma unroll
  for (int dt = 0; dt < 4; ++dt) {
#pragma unroll
    for (int v = 0; v < 8; ++v) {
      int d = dt * 16 + v + 8 * hf;                       // C: M = v + 8*half
      whn[((size_t)((b * H + h) * D + d)) * N + j0 + n] = (bf16_t)(acc[dt][v] * inv);
    }
  }
}

// ---- K4: h'[b,h,i,d] = sum_j P[i,j]*whn[j,d]; P generated on the fly; ELU; store transposed
__global__ void k_attn(const int* __restrict__ adj, const float* __restrict__ e1,
                       const float* __restrict__ e2, const bf16_t* __restrict__ whn,
                       float* __restrict__ out) {
  __shared__ __align__(16) int adjS[16 * 32];
  __shared__ float e2S[H][32];
  int bid = blockIdx.x;                 // B*(N/16) = 512, wave per head
  int b = bid >> 7; int it = bid & 127; int i0 = it * 16;
  int tid = threadIdx.x;
  int h = tid >> 5; int lane = tid & 31;
  int m = lane & 15; int hf = (lane >> 4) & 1;
  int row = tid >> 3, quad = tid & 7;   // cooperative adj tile loaders
  float e1v = e1[((size_t)(b * H + h)) * N + i0 + m];
  v8f acc[4] = {};
  for (int jc = 0; jc < N / 32; ++jc) {
    int j0 = jc * 32;
    __syncthreads();
    ((int4*)adjS)[tid] =
        *((const int4*)(adj + ((size_t)b * N + i0 + row) * N + j0) + quad);
    e2S[tid >> 5][tid & 31] =
        e2[((size_t)(b * H + (tid >> 5))) * N + j0 + (tid & 31)];
    __syncthreads();
    // Build A tile (P) directly in WMMA register layout
    v16bf av;
#pragma unroll
    for (int e = 0; e < 16; ++e) {
      int t = e >> 1, r = e & 1;
      int K = 16 * hf + 2 * t + r;
      float vv = e1v + e2S[h][K];
      vv = vv > 0.f ? vv : ALPHA * vv;
      float p = (adjS[m * 32 + K] != 0) ? __expf(vv) : 0.f;
      av[e] = (bf16_t)p;
    }
#pragma unroll
    for (int dt = 0; dt < 4; ++dt) {
      v16bf bvec = *(const v16bf*)(whn +
          ((size_t)((b * H + h) * D + dt * 16 + m)) * N + j0 + 16 * hf);
      acc[dt] = __builtin_amdgcn_wmma_f32_16x16x32_bf16(
          false, av, false, bvec, (short)0, acc[dt], false, false);
    }
  }
#pragma unroll
  for (int dt = 0; dt < 4; ++dt) {
#pragma unroll
    for (int v = 0; v < 8; ++v) {
      int i = i0 + v + 8 * hf;
      float val = acc[dt][v];
      float o = val > 0.f ? val : (__expf(val) - 1.f);    // ELU
      out[((size_t)b * N + i) * (H * D) + h * D + dt * 16 + m] = o;
    }
  }
}

extern "C" void kernel_launch(void* const* d_in, const int* in_sizes, int n_in,
                              void* d_out, int out_size, void* d_ws, size_t ws_size,
                              hipStream_t stream) {
  const float* x    = (const float*)d_in[0];
  const int*   adj  = (const int*)d_in[1];
  const float* oneh = (const float*)d_in[2];
  const float* Wq   = (const float*)d_in[3];
  const float* Wk   = (const float*)d_in[4];
  const float* W    = (const float*)d_in[5];
  const float* a    = (const float*)d_in[6];
  float* out = (float*)d_out;

  char* ws = (char*)d_ws;
  size_t o = 0;
  float*  vq    = (float*)(ws + o);  o += 4352;
  float*  vk    = (float*)(ws + o);  o += 4352;
  float*  e1    = (float*)(ws + o);  o += (size_t)B * H * N * 4;
  float*  e2    = (float*)(ws + o);  o += (size_t)B * H * N * 4;
  float*  denom = (float*)(ws + o);  o += (size_t)B * H * N * 4;
  bf16_t* Wt    = (bf16_t*)(ws + o); o += (size_t)H * D * F * 2;
  bf16_t* xb    = (bf16_t*)(ws + o); o += (size_t)B * N * F * 2;
  bf16_t* whn   = (bf16_t*)(ws + o); o += (size_t)B * H * D * N * 2;

  hipMemsetAsync(denom, 0, (size_t)B * H * N * 4, stream);
  k_vqvk<<<dim3((H * FT + 255) / 256), 256, 0, stream>>>(Wq, Wk, a, vq, vk);
  k_wt  <<<dim3(H * D * F / 256), 256, 0, stream>>>(W, Wt);
  k_xbf <<<dim3(B * N * F / 256), 256, 0, stream>>>(x, xb);
  k_e12 <<<dim3(B * H * N / 256), 256, 0, stream>>>(x, oneh, vq, vk, e1, e2);
  k_denom<<<dim3(B * 128), 256, 0, stream>>>(adj, e1, e2, denom);
  k_whn <<<dim3(B * H * 32), 128, 0, stream>>>(xb, Wt, denom, whn);
  k_attn<<<dim3(B * (N / 16)), 128, 0, stream>>>(adj, e1, e2, whn, out);
}
